// SingleHeadAlibiAttn_60842506715719
// MI455X (gfx1250) — matrix operand order repaired and checked by reading, not compiled, not run on hardware
//
#include <hip/hip_runtime.h>

#define BB 4
#define SS 2048
#define DD 1024
#define SCALE 0.03125f   // 1/sqrt(1024)

typedef __attribute__((ext_vector_type(16))) __bf16 v16bf;
typedef __attribute__((ext_vector_type(8)))  float  v8f;

union ABfrag {
    v16bf v;
    uint4 q[2];
    unsigned short s[16];
};

static __device__ __forceinline__ unsigned short f2bf_bits(float f) {
    union { float f; unsigned int u; } v; v.f = f;
    unsigned int r = v.u + 0x7FFFu + ((v.u >> 16) & 1u);   // RNE
    return (unsigned short)(r >> 16);
}

static __device__ __forceinline__ v8f zero8() {
    v8f z;
#pragma unroll
    for (int i = 0; i < 8; ++i) z[i] = 0.0f;
    return z;
}

static __device__ __forceinline__ v8f wmma_bf16(const v16bf& a, const v16bf& b, v8f c) {
    return __builtin_amdgcn_wmma_f32_16x16x32_bf16(false, a, false, b, (short)0, c, false, false);
}

// ---------------------------------------------------------------------------
// Prep kernels: f32 -> bf16 conversion (and transposed-convert for W_qk)
// ---------------------------------------------------------------------------
__global__ void cvt_bf16_kernel(const float* __restrict__ in,
                                unsigned short* __restrict__ out, int n) {
    for (int i = blockIdx.x * blockDim.x + threadIdx.x; i < n;
         i += gridDim.x * blockDim.x)
        out[i] = f2bf_bits(in[i]);
}

__global__ void transpose_cvt_kernel(const float* __restrict__ in,
                                     unsigned short* __restrict__ out) {
    // in: [DD][DD] row-major; out[c][r] = bf16(in[r][c])
    for (int i = blockIdx.x * blockDim.x + threadIdx.x; i < DD * DD;
         i += gridDim.x * blockDim.x) {
        int r = i / DD, c = i % DD;
        out[(size_t)c * DD + r] = f2bf_bits(in[i]);
    }
}

// ---------------------------------------------------------------------------
// GEMM core: each wave computes a 32x64 output tile (2 M-frags x 4 N-frags,
// 8 f32 accumulators). Per 32-K step: 2 A-frag + 4 B-frag loads feed 8 WMMAs
// (24 B/lane/WMMA operand traffic). Both operand matrices are stored so that
// every fragment is two contiguous 16B loads per lane.
//   A: rows of xb       [b*S+s][e]
//   B: rows of w        [n][e]      (weight pre-transposed as needed)
// grid: (B*S/32, D/512), block 256 (8 waves); wave w -> n0 = y*512 + w*64
// ---------------------------------------------------------------------------
template <bool STORE_TRANSPOSED>
static __device__ __forceinline__ void gemm_body(
    const unsigned short* __restrict__ xb,
    const unsigned short* __restrict__ wmat,
    unsigned short* __restrict__ outp) {
    const int tid  = threadIdx.x;
    const int wave = tid >> 5;
    const int lane = tid & 31;
    const int half = lane >> 4;
    const int r16  = lane & 15;
    const int s0   = blockIdx.x * 32;                 // global row (b*S+s)
    const int n0   = blockIdx.y * 512 + wave * 64;

    const unsigned short* arow0 = xb + ((size_t)(s0 + r16)) * DD;
    const unsigned short* arow1 = arow0 + (size_t)16 * DD;
    const unsigned short* brow  = wmat + ((size_t)(n0 + r16)) * DD;

    v8f acc[2][4];
#pragma unroll
    for (int mi = 0; mi < 2; ++mi)
#pragma unroll
        for (int ni = 0; ni < 4; ++ni) acc[mi][ni] = zero8();

    for (int e0 = 0; e0 < DD; e0 += 32) {
        ABfrag a0, a1;
        a0.q[0] = *(const uint4*)(arow0 + e0 + half * 16);
        a0.q[1] = *(const uint4*)(arow0 + e0 + half * 16 + 8);
        a1.q[0] = *(const uint4*)(arow1 + e0 + half * 16);
        a1.q[1] = *(const uint4*)(arow1 + e0 + half * 16 + 8);
#pragma unroll
        for (int ni = 0; ni < 4; ++ni) {
            ABfrag bf;
            const unsigned short* bp = brow + (size_t)ni * 16 * DD + e0 + half * 16;
            bf.q[0] = *(const uint4*)(bp);
            bf.q[1] = *(const uint4*)(bp + 8);
            acc[0][ni] = wmma_bf16(a0.v, bf.v, acc[0][ni]);
            acc[1][ni] = wmma_bf16(a1.v, bf.v, acc[1][ni]);
        }
    }

    if (!STORE_TRANSPOSED) {
        // row-major bf16 [B*S][D]
#pragma unroll
        for (int mi = 0; mi < 2; ++mi)
#pragma unroll
            for (int ni = 0; ni < 4; ++ni)
#pragma unroll
                for (int j = 0; j < 8; ++j) {
                    const int m = j + 8 * half;
                    outp[((size_t)(s0 + mi * 16 + m)) * DD + n0 + ni * 16 + r16] =
                        f2bf_bits(acc[mi][ni][j]);
                }
    } else {
        // transposed bf16 [b][d][s]; each lane packs 8 consecutive s into 16B
        const int b  = s0 / SS;
        const int sl = s0 % SS;
#pragma unroll
        for (int mi = 0; mi < 2; ++mi)
#pragma unroll
            for (int ni = 0; ni < 4; ++ni) {
                unsigned short pack[8];
#pragma unroll
                for (int j = 0; j < 8; ++j) pack[j] = f2bf_bits(acc[mi][ni][j]);
                const int d = n0 + ni * 16 + r16;
                *(uint4*)(outp + ((size_t)b * DD + d) * SS + sl + mi * 16 + 8 * half) =
                    *(const uint4*)pack;
            }
    }
}

__global__ __launch_bounds__(256) void qgemm_kernel(
    const unsigned short* __restrict__ xb,
    const unsigned short* __restrict__ wt,
    unsigned short* __restrict__ Qb) {
    gemm_body<false>(xb, wt, Qb);
}

__global__ __launch_bounds__(256) void vgemm_kernel(
    const unsigned short* __restrict__ xb,
    const unsigned short* __restrict__ wv,
    unsigned short* __restrict__ Vt) {
    gemm_body<true>(xb, wv, Vt);
}

// ---------------------------------------------------------------------------
// Flash attention: 1 block = 16 queries, 8 waves.
//   QK^T: contraction split across waves (128 dims each), reduced in LDS.
//   Online softmax in f32. PV: each wave owns 128 output columns.
// ---------------------------------------------------------------------------
__global__ __launch_bounds__(256) void attn_kernel(
    const unsigned short* __restrict__ Qb,   // [B][S][D] bf16
    const unsigned short* __restrict__ Kb,   // xb [B][S][D] bf16
    const unsigned short* __restrict__ Vt,   // [B][D][S] bf16
    const float* __restrict__ slope_p,
    float* __restrict__ out) {               // [B][S][D] f32
    const int tid  = threadIdx.x;
    const int wave = tid >> 5;
    const int lane = tid & 31;
    const int half = lane >> 4;
    const int r16  = lane & 15;
    const int b    = blockIdx.x / (SS / 16);
    const int q0   = (blockIdx.x % (SS / 16)) * 16;
    const float slope = slope_p[0];

    __shared__ float slab[8][16 * 32];
    __shared__ float Sred[16 * 32];
    __shared__ float Pbuf[16 * 32];
    __shared__ float mrow[16], lrow[16], arow[16];

    if (tid < 16) { mrow[tid] = -3.0e38f; lrow[tid] = 0.0f; }
    __syncthreads();

    v8f o[8];
#pragma unroll
    for (int t = 0; t < 8; ++t) o[t] = zero8();

    const unsigned short* Qrow = Qb + ((size_t)b * SS + q0 + r16) * DD;
    const int e_base = wave * 128;
    const int kmax = q0 + 15;

    for (int k0 = 0; k0 <= kmax; k0 += 32) {
        // ---- partial QK^T over e in [e_base, e_base+128)
        v8f sc0 = zero8(), sc1 = zero8();
        for (int ec = 0; ec < 128; ec += 32) {
            const int e0 = e_base + ec;
            ABfrag a, b0, b1;
            a.q[0] = *(const uint4*)(Qrow + e0 + half * 16);
            a.q[1] = *(const uint4*)(Qrow + e0 + half * 16 + 8);
            const unsigned short* kr0 =
                Kb + ((size_t)b * SS + k0 + r16) * DD + e0 + half * 16;
            const unsigned short* kr1 = kr0 + (size_t)16 * DD;
            b0.q[0] = *(const uint4*)(kr0);
            b0.q[1] = *(const uint4*)(kr0 + 8);
            b1.q[0] = *(const uint4*)(kr1);
            b1.q[1] = *(const uint4*)(kr1 + 8);
            sc0 = wmma_bf16(a.v, b0.v, sc0);
            sc1 = wmma_bf16(a.v, b1.v, sc1);
        }
        {
            float* sl = slab[wave];
#pragma unroll
            for (int j = 0; j < 8; ++j) {
                const int m = j + 8 * half;
                sl[m * 32 + r16]      = sc0[j];
                sl[m * 32 + 16 + r16] = sc1[j];
            }
        }
        __syncthreads();

        // ---- reduce 8 slabs, scale, alibi bias, causal mask (512 elems / 256 thr)
        {
            int idx = tid;
#pragma unroll
            for (int t = 0; t < 2; ++t, idx += 256) {
                const int m = idx >> 5, n = idx & 31;
                float acc = 0.0f;
#pragma unroll
                for (int w = 0; w < 8; ++w) acc += slab[w][idx];
                const int kk = k0 + n, qq = q0 + m;
                float s = acc * SCALE + (float)(kk - qq) * slope;
                Sred[idx] = (kk <= qq) ? s : -3.0e38f;
            }
        }
        __syncthreads();

        // ---- per-row online softmax (16 rows, threads 0..15)
        if (tid < 16) {
            const int m = tid;
            float rowv[32];
            float mx = -3.0e38f;
#pragma unroll
            for (int n = 0; n < 32; ++n) {
                rowv[n] = Sred[m * 32 + n];
                mx = fmaxf(mx, rowv[n]);
            }
            const float mprev = mrow[m];
            const float mnew  = fmaxf(mprev, mx);
            const float alpha = __expf(mprev - mnew);
            float ls = 0.0f;
#pragma unroll
            for (int n = 0; n < 32; ++n) {
                const float p = __expf(rowv[n] - mnew);
                Pbuf[m * 32 + n] = p;
                ls += p;
            }
            lrow[m] = lrow[m] * alpha + ls;
            mrow[m] = mnew;
            arow[m] = alpha;
        }
        __syncthreads();

        // ---- rescale O, build P A-frag, PV WMMAs over this wave's 128 columns
        float al[8];
#pragma unroll
        for (int j = 0; j < 8; ++j) al[j] = arow[j + 8 * half];

        ABfrag pa;
#pragma unroll
        for (int i = 0; i < 8; ++i) {
            pa.s[i]     = f2bf_bits(Pbuf[r16 * 32 + half * 16 + i]);
            pa.s[i + 8] = f2bf_bits(Pbuf[r16 * 32 + half * 16 + 8 + i]);
        }
        const unsigned short* vbase =
            Vt + ((size_t)b * DD + wave * 128 + r16) * SS + k0 + half * 16;
#pragma unroll
        for (int t = 0; t < 8; ++t) {
#pragma unroll
            for (int j = 0; j < 8; ++j) o[t][j] *= al[j];
            ABfrag vb;
            const unsigned short* vp = vbase + (size_t)t * 16 * SS;
            vb.q[0] = *(const uint4*)(vp);
            vb.q[1] = *(const uint4*)(vp + 8);
            o[t] = wmma_bf16(pa.v, vb.v, o[t]);
        }
        __syncthreads();
    }

    // ---- normalize by row sums and store f32
    float li[8];
#pragma unroll
    for (int j = 0; j < 8; ++j) li[j] = 1.0f / lrow[j + 8 * half];
#pragma unroll
    for (int t = 0; t < 8; ++t) {
        const int dcol = wave * 128 + t * 16 + r16;
#pragma unroll
        for (int j = 0; j < 8; ++j) {
            const int m = j + 8 * half;
            out[((size_t)b * SS + q0 + m) * DD + dcol] = o[t][j] * li[j];
        }
    }
}

// ---------------------------------------------------------------------------
extern "C" void kernel_launch(void* const* d_in, const int* in_sizes, int n_in,
                              void* d_out, int out_size, void* d_ws, size_t ws_size,
                              hipStream_t stream) {
    (void)in_sizes; (void)n_in; (void)out_size; (void)ws_size;
    const float* x     = (const float*)d_in[0];
    const float* W_qk  = (const float*)d_in[1];
    const float* W_v   = (const float*)d_in[2];
    const float* slope = (const float*)d_in[3];
    float* out = (float*)d_out;

    const size_t NX = (size_t)BB * SS * DD;   // 8,388,608
    const size_t NW = (size_t)DD * DD;        // 1,048,576

    unsigned short* xb  = (unsigned short*)d_ws;
    unsigned short* Qb  = xb + NX;
    unsigned short* Vt  = Qb + NX;
    unsigned short* Wt  = Vt + NX;            // W_qk transposed bf16
    unsigned short* Wvb = Wt + NW;            // W_v bf16

    cvt_bf16_kernel<<<4096, 256, 0, stream>>>(x, xb, (int)NX);
    transpose_cvt_kernel<<<2048, 256, 0, stream>>>(W_qk, Wt);
    cvt_bf16_kernel<<<2048, 256, 0, stream>>>(W_v, Wvb, (int)NW);

    dim3 gg(BB * SS / 32, DD / 512);          // (256, 2)
    qgemm_kernel<<<gg, 256, 0, stream>>>(xb, Wt, Qb);
    vgemm_kernel<<<gg, 256, 0, stream>>>(xb, Wvb, Vt);

    attn_kernel<<<BB * SS / 16, 256, 0, stream>>>(Qb, xb, Vt, slope, out);
}
